// MoE_63127429317119
// MI455X (gfx1250) — compile-verified, matrix operand order checked by reading, not run on hardware
//
#include <hip/hip_runtime.h>
#include <hip/hip_bf16.h>

// ---------------- problem constants ----------------
constexpr int T_TOK = 8192;
constexpr int D_DIM = 1024;
constexpr int F_DIM = 4096;
constexpr int E_NUM = 8;
constexpr int CAP   = 1024;

typedef __bf16 bf16;
typedef __attribute__((ext_vector_type(16))) __bf16 v16bf;
typedef __attribute__((ext_vector_type(8)))  __bf16 v8bf;
typedef __attribute__((ext_vector_type(8)))  float  v8f;

// ---------------- WMMA helpers ----------------
__device__ __forceinline__ v8f wmma_bf16(v16bf a, v16bf b, v8f c) {
    // D = A(16x32 bf16) x B(32x16 bf16) + C(16x16 f32)
    return __builtin_amdgcn_wmma_f32_16x16x32_bf16(
        /*neg_a=*/false, a, /*neg_b=*/false, b,
        /*c_mod=*/(short)0, c, /*reuse_a=*/false, /*reuse_b=*/false);
}

// Load one 16-bf16-per-lane A/B fragment. `p` = row/col base + k0 + half*8.
// ISA 16-bit A layout: per lane, VGPR0..3 hold K = half*8 + {0..7},
// VGPR4..7 hold K = 16 + half*8 + {0..7} -> two contiguous 16-byte chunks.
__device__ __forceinline__ v16bf load_frag16(const bf16* p) {
    v8bf lo = *(const v8bf*)(p);
    v8bf hi = *(const v8bf*)(p + 16);
    return __builtin_shufflevector(lo, hi,
        0,1,2,3,4,5,6,7,8,9,10,11,12,13,14,15);
}

// ---------------- 0) zero the output ----------------
__global__ void __launch_bounds__(256)
zero_out_kernel(float* __restrict__ out) {
    size_t i = (size_t)blockIdx.x * 256 + threadIdx.x;
    out[i] = 0.0f;
}

// ---------------- 1) router: logits, softmax, top-1 ----------------
__global__ void __launch_bounds__(256)
router_kernel(const float* __restrict__ x, const float* __restrict__ Wr,
              int* __restrict__ top_e, float* __restrict__ top_w) {
    __shared__ float sWr[D_DIM * E_NUM];          // 32 KB of the 320 KB LDS
    for (int i = threadIdx.x; i < D_DIM * E_NUM; i += 256) sWr[i] = Wr[i];
    __syncthreads();

    const int t = blockIdx.x * 256 + threadIdx.x;  // T_TOK % 256 == 0
    const float* xr = x + (size_t)t * D_DIM;
    float logit[E_NUM];
#pragma unroll
    for (int e = 0; e < E_NUM; ++e) logit[e] = 0.0f;
    for (int d = 0; d < D_DIM; ++d) {
        const float xv = xr[d];
#pragma unroll
        for (int e = 0; e < E_NUM; ++e)
            logit[e] = fmaf(xv, sWr[d * E_NUM + e], logit[e]);
    }
    float m = logit[0]; int best = 0;
#pragma unroll
    for (int e = 1; e < E_NUM; ++e)
        if (logit[e] > m) { m = logit[e]; best = e; }   // first max wins, like argmax
    float sum = 0.0f;
#pragma unroll
    for (int e = 0; e < E_NUM; ++e) sum += __expf(logit[e] - m);
    top_e[t] = best;
    top_w[t] = 1.0f / sum;                         // probs[argmax] = exp(0)/sum
}

// ---------------- 2) ordered dispatch (matches jnp.nonzero ordering) ----------------
__global__ void __launch_bounds__(1024)
dispatch_kernel(const int* __restrict__ top_e, int* __restrict__ idx) {
    __shared__ int s[1024];
    const int e = blockIdx.x;
    const int tid = threadIdx.x;
    int running = 0;
    for (int chunk = 0; chunk < T_TOK / 1024; ++chunk) {
        const int t = chunk * 1024 + tid;
        const int m = (top_e[t] == e) ? 1 : 0;
        s[tid] = m;
        __syncthreads();
        // Hillis-Steele inclusive scan over 1024 entries
        for (int off = 1; off < 1024; off <<= 1) {
            int v = (tid >= off) ? s[tid - off] : 0;
            __syncthreads();
            s[tid] += v;
            __syncthreads();
        }
        if (m) {
            const int pos = running + s[tid] - 1;
            if (pos < CAP) idx[e * CAP + pos] = t;  // ascending token order
        }
        running += s[1023];
        __syncthreads();
    }
    // pad remaining capacity slots with T (OOB sentinel, dropped on scatter)
    for (int p = running + tid; p < CAP; p += 1024) idx[e * CAP + p] = T_TOK;
}

// ---------------- 3) gather + f32->bf16 convert ----------------
__global__ void __launch_bounds__(256)
gather_kernel(const float* __restrict__ x, const int* __restrict__ idx,
              bf16* __restrict__ Xg) {
    const int slot = blockIdx.x;                   // e*CAP + c
    const int t = idx[slot];
    bf16* dst = Xg + (size_t)slot * D_DIM;
    if (t < T_TOK) {
        const float* src = x + (size_t)t * D_DIM;
        for (int d = threadIdx.x; d < D_DIM; d += 256) dst[d] = (bf16)src[d];
    } else {
        for (int d = threadIdx.x; d < D_DIM; d += 256) dst[d] = (bf16)0.0f;
    }
}

// ---------------- 4) transpose + convert weights: src[R][C] f32 -> dst[C][R] bf16 ----
__global__ void __launch_bounds__(256)
transpose_cvt_kernel(const float* __restrict__ src, bf16* __restrict__ dst,
                     int R, int C) {
    __shared__ float tile[32][33];
    const int e = blockIdx.z;
    src += (size_t)e * R * C;
    dst += (size_t)e * R * C;
    const int c0 = blockIdx.x * 32;
    const int r0 = blockIdx.y * 32;
    const int tx = threadIdx.x;
    for (int i = threadIdx.y; i < 32; i += 8)
        tile[i][tx] = src[(size_t)(r0 + i) * C + c0 + tx];
    __syncthreads();
    for (int i = threadIdx.y; i < 32; i += 8)
        dst[(size_t)(c0 + i) * R + r0 + tx] = (bf16)tile[tx][i];
}

// ---------------- 5) GEMM1: hidden = relu(Xg @ W1 + b1), bf16 out ----------------
// A: Xg[e] (CAP x D, bf16 rowmajor). B: W1T[e] (F x D, bf16 rowmajor == W1^T).
// Block = 4 waves in a 2x2 grid; each wave owns a 64x64 patch (4x4 WMMA tiles).
// Per K-step: 16 x global_load_b128 feed 16 x v_wmma (1:1 issue ratio, 32 FLOP/B).
__global__ void __launch_bounds__(128)
gemm1_kernel(const bf16* __restrict__ Xg, const bf16* __restrict__ W1T,
             const float* __restrict__ b1, bf16* __restrict__ H) {
    const int e = blockIdx.z;
    const bf16* A = Xg + (size_t)e * CAP * D_DIM;
    const bf16* B = W1T + (size_t)e * (size_t)F_DIM * D_DIM;
    const float* bias = b1 + (size_t)e * F_DIM;
    bf16* C = H + (size_t)e * (size_t)CAP * F_DIM;

    const int tid  = threadIdx.x;
    const int lane = tid & 31;
    const int w    = tid >> 5;        // 4 waves
    const int wr   = w >> 1, wc = w & 1;
    const int l16  = lane & 15, half = lane >> 4;

    const int rowBase = blockIdx.y * 128 + wr * 64;
    const int colBase = blockIdx.x * 128 + wc * 64;

    v8f acc[4][4] = {};
    const bf16* Arow[4];
    const bf16* Brow[4];
#pragma unroll
    for (int i = 0; i < 4; ++i) {
        Arow[i] = A + (size_t)(rowBase + i * 16 + l16) * D_DIM + half * 8;
        Brow[i] = B + (size_t)(colBase + i * 16 + l16) * D_DIM + half * 8;
    }

    for (int k0 = 0; k0 < D_DIM; k0 += 32) {
        v16bf a[4], b[4];
#pragma unroll
        for (int i = 0; i < 4; ++i) {
            a[i] = load_frag16(Arow[i] + k0);
            b[i] = load_frag16(Brow[i] + k0);
        }
#pragma unroll
        for (int i = 0; i < 4; ++i)
#pragma unroll
            for (int j = 0; j < 4; ++j)
                acc[i][j] = wmma_bf16(a[i], b[j], acc[i][j]);
    }

#pragma unroll
    for (int i = 0; i < 4; ++i) {
#pragma unroll
        for (int j = 0; j < 4; ++j) {
            const int col = colBase + j * 16 + l16;
            const float bv = bias[col];
#pragma unroll
            for (int v = 0; v < 8; ++v) {
                const int row = rowBase + i * 16 + half * 8 + v;
                float val = acc[i][j][v] + bv;
                val = fmaxf(val, 0.0f);
                C[(size_t)row * F_DIM + col] = (bf16)val;
            }
        }
    }
}

// ---------------- 6) GEMM2 + fused scatter: out[idx] = (H @ W2 + b2) * top_w ------
// A: H[e] (CAP x F bf16). B: W2T[e] (D x F bf16 == W2^T).
__global__ void __launch_bounds__(128)
gemm2_kernel(const bf16* __restrict__ H, const bf16* __restrict__ W2T,
             const float* __restrict__ b2, const int* __restrict__ idx,
             const float* __restrict__ top_w, float* __restrict__ out) {
    const int e = blockIdx.z;
    const bf16* A = H + (size_t)e * (size_t)CAP * F_DIM;
    const bf16* B = W2T + (size_t)e * (size_t)D_DIM * F_DIM;
    const float* bias = b2 + (size_t)e * D_DIM;
    const int* eidx = idx + e * CAP;

    const int tid  = threadIdx.x;
    const int lane = tid & 31;
    const int w    = tid >> 5;
    const int wr   = w >> 1, wc = w & 1;
    const int l16  = lane & 15, half = lane >> 4;

    const int rowBase = blockIdx.y * 128 + wr * 64;   // slot within expert
    const int colBase = blockIdx.x * 128 + wc * 64;   // model dim

    v8f acc[4][4] = {};
    const bf16* Arow[4];
    const bf16* Brow[4];
#pragma unroll
    for (int i = 0; i < 4; ++i) {
        Arow[i] = A + (size_t)(rowBase + i * 16 + l16) * F_DIM + half * 8;
        Brow[i] = B + (size_t)(colBase + i * 16 + l16) * F_DIM + half * 8;
    }

    for (int k0 = 0; k0 < F_DIM; k0 += 32) {
        v16bf a[4], b[4];
#pragma unroll
        for (int i = 0; i < 4; ++i) {
            a[i] = load_frag16(Arow[i] + k0);
            b[i] = load_frag16(Brow[i] + k0);
        }
#pragma unroll
        for (int i = 0; i < 4; ++i)
#pragma unroll
            for (int j = 0; j < 4; ++j)
                acc[i][j] = wmma_bf16(a[i], b[j], acc[i][j]);
    }

#pragma unroll
    for (int i = 0; i < 4; ++i) {
#pragma unroll
        for (int j = 0; j < 4; ++j) {
            const int col = colBase + j * 16 + l16;
            const float bv = bias[col];
#pragma unroll
            for (int v = 0; v < 8; ++v) {
                const int c = rowBase + i * 16 + half * 8 + v;
                const int t = eidx[c];
                if (t < T_TOK) {
                    out[(size_t)t * D_DIM + col] = (acc[i][j][v] + bv) * top_w[t];
                }
            }
        }
    }
}

// ---------------- workspace layout (bytes, 256-aligned) ----------------
constexpr size_t OFF_XG   = 0;                                   // E*CAP*D bf16 = 16 MB
constexpr size_t OFF_W1T  = OFF_XG  + (size_t)E_NUM * CAP * D_DIM * 2;          // 64 MB
constexpr size_t OFF_W2T  = OFF_W1T + (size_t)E_NUM * F_DIM * D_DIM * 2;        // 64 MB
constexpr size_t OFF_H    = OFF_W2T + (size_t)E_NUM * D_DIM * F_DIM * 2;        // 64 MB
constexpr size_t OFF_IDX  = OFF_H   + (size_t)E_NUM * CAP * F_DIM * 2;
constexpr size_t OFF_TOPE = OFF_IDX  + (size_t)E_NUM * CAP * 4;
constexpr size_t OFF_TOPW = OFF_TOPE + (size_t)T_TOK * 4;

extern "C" void kernel_launch(void* const* d_in, const int* in_sizes, int n_in,
                              void* d_out, int out_size, void* d_ws, size_t ws_size,
                              hipStream_t stream) {
    const float* x  = (const float*)d_in[0];
    const float* Wr = (const float*)d_in[1];
    const float* W1 = (const float*)d_in[2];
    const float* b1 = (const float*)d_in[3];
    const float* W2 = (const float*)d_in[4];
    const float* b2 = (const float*)d_in[5];
    float* out = (float*)d_out;

    char* ws = (char*)d_ws;
    bf16*  Xg    = (bf16*)(ws + OFF_XG);
    bf16*  W1T   = (bf16*)(ws + OFF_W1T);
    bf16*  W2T   = (bf16*)(ws + OFF_W2T);
    bf16*  Hbuf  = (bf16*)(ws + OFF_H);
    int*   idx   = (int*)(ws + OFF_IDX);
    int*   top_e = (int*)(ws + OFF_TOPE);
    float* top_w = (float*)(ws + OFF_TOPW);

    // 0) zero output (capacity-dropped tokens must read 0)
    zero_out_kernel<<<(T_TOK * D_DIM) / 256, 256, 0, stream>>>(out);
    // 1) router
    router_kernel<<<T_TOK / 256, 256, 0, stream>>>(x, Wr, top_e, top_w);
    // 2) ordered dispatch with capacity clipping
    dispatch_kernel<<<E_NUM, 1024, 0, stream>>>(top_e, idx);
    // 3) gather tokens -> bf16
    gather_kernel<<<E_NUM * CAP, 256, 0, stream>>>(x, idx, Xg);
    // 4) weight transpose+convert (B operands want contiguous-K bf16 rows)
    transpose_cvt_kernel<<<dim3(F_DIM / 32, D_DIM / 32, E_NUM), dim3(32, 8), 0, stream>>>(
        W1, W1T, D_DIM, F_DIM);
    transpose_cvt_kernel<<<dim3(D_DIM / 32, F_DIM / 32, E_NUM), dim3(32, 8), 0, stream>>>(
        W2, W2T, F_DIM, D_DIM);
    // 5) expert GEMM1 (+bias, relu)
    gemm1_kernel<<<dim3(F_DIM / 128, CAP / 128, E_NUM), 128, 0, stream>>>(Xg, W1T, b1, Hbuf);
    // 6) expert GEMM2 (+bias) fused with weighted scatter
    gemm2_kernel<<<dim3(D_DIM / 128, CAP / 128, E_NUM), 128, 0, stream>>>(
        Hbuf, W2T, b2, idx, top_w, out);
}
